// DGCNNSemSeg_78365973283369
// MI455X (gfx1250) — compile-verified, matrix operand order checked by reading, not run on hardware
//
#include <hip/hip_runtime.h>
#include <hip/hip_bf16.h>
#include <stdint.h>

// ---------------------------------------------------------------------------
// DGCNN semantic segmentation for MI455X (gfx1250, wave32, WMMA).
// GEMM math -> v_wmma_f32_16x16x32_f16 (f16 in, f32 accumulate).
// GEMM A-tiles staged to LDS via gfx1250 async global->LDS copies.
// ---------------------------------------------------------------------------

typedef __attribute__((ext_vector_type(16))) _Float16 v16h;
typedef __attribute__((ext_vector_type(8)))  float    v8f;
typedef __attribute__((ext_vector_type(4)))  int      v4i;
typedef __attribute__((address_space(1))) v4i gv4i;   // global int4
typedef __attribute__((address_space(3))) v4i lv4i;   // LDS int4

#define NPTS   4096
#define BATCH  4
#define KNN    20
#define LEAK   0.2f
#define BNEPS  1e-5f

#if defined(__has_builtin)
#if __has_builtin(__builtin_amdgcn_global_load_async_to_lds_b128)
#define HAVE_ASYNC_LDS 1
#endif
#endif

// K index inside a 16-bit WMMA A/B fragment (ISA 7.12.2, 16-bit A 16x32):
// lanes 0-15: h=0..7 -> K=0..7, h=8..15 -> K=16..23; lanes 16-31: +8.
__device__ __forceinline__ int kmapf(int lane, int h) {
  return (h & 7) + ((h >> 3) << 4) + ((lane >> 4) << 3);
}

__device__ __forceinline__ float lrelu(float v) { return v >= 0.f ? v : LEAK * v; }

__device__ __forceinline__ void atomicMaxF(float* addr, float val) {
  int* ai = reinterpret_cast<int*>(addr);
  int old = __float_as_int(*addr);
  while (__int_as_float(old) < val) {
    int prev = atomicCAS(ai, old, __float_as_int(val));
    if (prev == old) break;
    old = prev;
  }
}

// 16-byte global -> LDS copy (per-lane). Async on gfx1250 toolchains that
// expose the builtin; falls back to a vectorized through-VGPR copy.
__device__ __forceinline__ void copy16B_g2l(_Float16* ldst, const _Float16* gsrc) {
#if defined(HAVE_ASYNC_LDS)
  __builtin_amdgcn_global_load_async_to_lds_b128((gv4i*)gsrc, (lv4i*)ldst, 0, 0);
#else
  *(uint4*)ldst = *(const uint4*)gsrc;
#endif
}

__device__ __forceinline__ void wait_async_copies() {
#if defined(HAVE_ASYNC_LDS)
#if __has_builtin(__builtin_amdgcn_s_wait_asynccnt)
  __builtin_amdgcn_s_wait_asynccnt(0);
#else
  asm volatile("s_wait_asynccnt 0x0" ::: "memory");
#endif
#endif
}

// ---------------------------------------------------------------------------
// Pack a KxN fp32 weight matrix into f16 WMMA B-fragments.
// frag (nt,kt) = 32 lanes x 16 halfs; element (lane,h) = W[k][n],
// k = kt*32 + kmapf(lane,h), n = nt*16 + (lane&15). Zero-padded OOB.
// ---------------------------------------------------------------------------
__global__ void pack_w_kernel(const float* __restrict__ W, _Float16* __restrict__ Wp,
                              int K, int Nreal, int KT, int NT) {
  int t = blockIdx.x * blockDim.x + threadIdx.x;
  int total = NT * KT * 32;
  if (t >= total) return;
  int lane = t & 31;
  int tile = t >> 5;
  int kt = tile % KT;
  int ntl = tile / KT;
  #pragma unroll
  for (int h = 0; h < 16; ++h) {
    int k = kt * 32 + kmapf(lane, h);
    int n = ntl * 16 + (lane & 15);
    float v = (k < K && n < Nreal) ? W[(size_t)k * Nreal + n] : 0.f;
    Wp[(size_t)t * 16 + h] = (_Float16)v;
  }
}

__global__ void fill_kernel(float* p, float v, int n) {
  int i = blockIdx.x * blockDim.x + threadIdx.x;
  if (i < n) p[i] = v;
}

__global__ void f32_to_f16_kernel(const float* __restrict__ s,
                                  _Float16* __restrict__ d, int n) {
  int i = blockIdx.x * blockDim.x + threadIdx.x;
  if (i < n) d[i] = (_Float16)s[i];
}

// ---------------------------------------------------------------------------
// knn: one thread per query point. neg squared distance, sorted top-20
// (strict-greater insertion reproduces top_k first-occurrence tie order).
// ---------------------------------------------------------------------------
template <int C>
__global__ void knn_kernel(const float* __restrict__ x, int stride,
                           int* __restrict__ idx) {
  int t = blockIdx.x * blockDim.x + threadIdx.x;
  if (t >= BATCH * NPTS) return;
  int b = t >> 12, n = t & (NPTS - 1);
  const float* xb = x + (size_t)b * NPTS * stride;
  float q[C];
  #pragma unroll
  for (int c = 0; c < C; ++c) q[c] = xb[(size_t)n * stride + c];

  float best[KNN];
  int   bidx[KNN];
  #pragma unroll
  for (int j = 0; j < KNN; ++j) { best[j] = -3.4e38f; bidx[j] = 0; }

  for (int m = 0; m < NPTS; ++m) {
    const float* pm = xb + (size_t)m * stride;
    float d = 0.f;
    #pragma unroll
    for (int c = 0; c < C; ++c) { float df = q[c] - pm[c]; d -= df * df; }
    if (d > best[KNN - 1]) {
      int j = KNN - 1;
      while (j > 0 && d > best[j - 1]) {
        best[j] = best[j - 1]; bidx[j] = bidx[j - 1]; --j;
      }
      best[j] = d; bidx[j] = m;
    }
  }
  #pragma unroll
  for (int j = 0; j < KNN; ++j) idx[(size_t)t * KNN + j] = bidx[j];
}

// ---------------------------------------------------------------------------
// EdgeConv: one wave per point. Edge features [ctr, nbr-ctr] (20 rows padded
// to 32) staged in LDS as f16; WMMA over 2 M-tiles x COUT/16 N-tiles x
// 2C/32 K-tiles; epilogue fuses BN + LeakyReLU + masked max over k and
// writes both fp32 (for the next knn) and f16 (for the WMMA GEMMs).
// ---------------------------------------------------------------------------
template <int CIN, int COUT>
__global__ __launch_bounds__(128) void edgeconv_kernel(
    const float* __restrict__ x, int xstride, const int* __restrict__ idx,
    const _Float16* __restrict__ Wp,
    const float* __restrict__ gamma, const float* __restrict__ beta,
    const float* __restrict__ mean,  const float* __restrict__ var,
    float* __restrict__ y, _Float16* __restrict__ yh, int ldo, int coloff) {
  constexpr int K2  = 2 * CIN;
  constexpr int K2P = (K2 + 31) & ~31;
  constexpr int KT  = K2P / 32;
  constexpr int NT  = COUT / 16;

  __shared__ _Float16 sA[4][32 * K2P];

  const int w = threadIdx.x >> 5;
  const int lane = threadIdx.x & 31;
  const int p = blockIdx.x * 4 + w;            // point id, grid covers B*N
  const int b = p >> 12;
  const int n = p & (NPTS - 1);
  const float* xb = x + (size_t)b * NPTS * xstride;
  _Float16* A = sA[w];

  for (int i = lane; i < 32 * K2P; i += 32) A[i] = (_Float16)0.f;
  __syncthreads();

  const float* pc = xb + (size_t)n * xstride;
  for (int kk = 0; kk < KNN; ++kk) {
    int nb = idx[(size_t)p * KNN + kk];
    const float* pn = xb + (size_t)nb * xstride;
    for (int c = lane; c < CIN; c += 32) {
      float ctr = pc[c];
      A[kk * K2P + c]       = (_Float16)ctr;
      A[kk * K2P + CIN + c] = (_Float16)(pn[c] - ctr);
    }
  }
  __syncthreads();

  const int r16 = lane & 15;
  for (int nt = 0; nt < NT; ++nt) {
    int col = nt * 16 + r16;
    float scale = gamma[col] * rsqrtf(var[col] + BNEPS);
    float shift = beta[col] - mean[col] * scale;

    v8f c0 = {}; v8f c1 = {};
    for (int kt = 0; kt < KT; ++kt) {
      v16h bf = *((const v16h*)(Wp + ((size_t)(nt * KT + kt) * 32 + lane) * 16));
      v16h a0, a1;
      #pragma unroll
      for (int h = 0; h < 16; ++h) {
        int k = kt * 32 + kmapf(lane, h);
        a0[h] = A[r16 * K2P + k];
        a1[h] = A[(16 + r16) * K2P + k];
      }
      c0 = __builtin_amdgcn_wmma_f32_16x16x32_f16(false, a0, false, bf,
                                                  (short)0, c0, false, false);
      c1 = __builtin_amdgcn_wmma_f32_16x16x32_f16(false, a1, false, bf,
                                                  (short)0, c1, false, false);
    }

    float m = -3.4e38f;
    #pragma unroll
    for (int r = 0; r < 8; ++r) {
      int row0 = r + ((lane >> 4) << 3);       // edge row (k-index) in tile 0
      float v0 = lrelu(c0[r] * scale + shift);
      if (row0 < KNN) m = fmaxf(m, v0);
      int row1 = 16 + row0;                    // tile 1 (rows 16..31)
      float v1 = lrelu(c1[r] * scale + shift);
      if (row1 < KNN) m = fmaxf(m, v1);
    }
    m = fmaxf(m, __shfl_xor(m, 16, 32));       // combine half-lane row groups
    if (lane < 16) {
      y [(size_t)p * ldo + coloff + nt * 16 + lane] = m;
      yh[(size_t)p * ldo + coloff + nt * 16 + lane] = (_Float16)m;
    }
  }
}

// ---------------------------------------------------------------------------
// Block-tiled WMMA GEMM: Y = post(A @ W), A already f16 in global.
// One block = 4 waves = one 16-row M-tile x 4 N-tiles. The 16xK A-tile is
// staged into LDS ONCE per block with async global->LDS b128 copies and
// shared by all four waves (and all their K-tiles).
// CONCAT via splitK: A(row,k) = k<splitK ? Ah[row][k] : A2h[row/4096][k-splitK]
// RMAX: epilogue max-reduces tile rows, atomic-max into Yf[b][col]
// OUTH: store f16 (feeds the next GEMM) instead of fp32
// ---------------------------------------------------------------------------
template <bool BN, bool RELU, bool RMAX, bool OUTH>
__global__ __launch_bounds__(128) void gemm_kernel(
    const _Float16* __restrict__ Ah, int lda,
    const _Float16* __restrict__ A2h, int lda2, int splitK,
    const _Float16* __restrict__ Wp, int K, int NT, int Nreal,
    const float* __restrict__ gamma, const float* __restrict__ beta,
    const float* __restrict__ mean,  const float* __restrict__ var,
    void* __restrict__ Yout, int ldo) {
  extern __shared__ char dyn_smem[];
  _Float16* sA = (_Float16*)dyn_smem;          // [16][K] halfs

  const int tid = threadIdx.x;
  const int w = tid >> 5;
  const int lane = tid & 31;
  const int mt = blockIdx.x;
  const int row0 = mt * 16;

  // ---- stage A tile: 16*K halfs = 2K 16-byte chunks (K % 8 == 0) ----
  const int chunks = (16 * K) >> 3;            // multiple of 128 for our K's
  for (int i = tid; i < chunks; i += 128) {
    int linear = i << 3;
    int r = linear / K;
    int k = linear - r * K;
    const _Float16* src = (k < splitK)
        ? Ah  + (size_t)(row0 + r) * lda  + k
        : A2h + (size_t)((row0 + r) >> 12) * lda2 + (k - splitK);
    copy16B_g2l(sA + linear, src);
  }
  wait_async_copies();
  __syncthreads();

  const int nt = blockIdx.y * 4 + w;
  if (nt < NT) {                               // wave-uniform
    const int KT = K >> 5;
    const int r16 = lane & 15;
    const int col = nt * 16 + r16;

    v8f acc = {};
    for (int kt = 0; kt < KT; ++kt) {
      v16h bf = *((const v16h*)(Wp + ((size_t)(nt * KT + kt) * 32 + lane) * 16));
      v16h a;
      #pragma unroll
      for (int h = 0; h < 16; ++h)
        a[h] = sA[r16 * K + kt * 32 + kmapf(lane, h)];
      acc = __builtin_amdgcn_wmma_f32_16x16x32_f16(false, a, false, bf,
                                                   (short)0, acc, false, false);
    }

    float scale = 1.f, shift = 0.f;
    if (BN && col < Nreal) {
      scale = gamma[col] * rsqrtf(var[col] + BNEPS);
      shift = beta[col] - mean[col] * scale;
    }

    if (RMAX) {
      float m = -3.4e38f;
      #pragma unroll
      for (int r = 0; r < 8; ++r) {
        float v = acc[r] * scale + shift;
        if (RELU) v = lrelu(v);
        m = fmaxf(m, v);
      }
      m = fmaxf(m, __shfl_xor(m, 16, 32));
      if (lane < 16)                           // 256 m-tiles per batch
        atomicMaxF((float*)Yout + (size_t)(mt >> 8) * ldo + col, m);
    } else {
      #pragma unroll
      for (int r = 0; r < 8; ++r) {
        float v = acc[r] * scale + shift;
        if (RELU) v = lrelu(v);
        int rr = row0 + r + ((lane >> 4) << 3);
        if (col < Nreal) {
          if (OUTH) ((_Float16*)Yout)[(size_t)rr * ldo + col] = (_Float16)v;
          else      ((float*)Yout)   [(size_t)rr * ldo + col] = v;
        }
      }
    }
  }
}

// ---------------------------------------------------------------------------
// Host orchestration
// ---------------------------------------------------------------------------
static inline size_t packed_halfs(int K, int N) {
  int KT = (K + 31) / 32, NT = (N + 15) / 16;
  return (size_t)NT * KT * 32 * 16;
}

extern "C" void kernel_launch(void* const* d_in, const int* in_sizes, int n_in,
                              void* d_out, int out_size, void* d_ws, size_t ws_size,
                              hipStream_t stream) {
  const float* x   = (const float*)d_in[0];
  const float* W1  = (const float*)d_in[1];
  const float* bn1g = (const float*)d_in[2],  *bn1b = (const float*)d_in[3];
  const float* bn1m = (const float*)d_in[4],  *bn1v = (const float*)d_in[5];
  const float* W2  = (const float*)d_in[6];
  const float* bn2g = (const float*)d_in[7],  *bn2b = (const float*)d_in[8];
  const float* bn2m = (const float*)d_in[9],  *bn2v = (const float*)d_in[10];
  const float* W3  = (const float*)d_in[11];
  const float* bn3g = (const float*)d_in[12], *bn3b = (const float*)d_in[13];
  const float* bn3m = (const float*)d_in[14], *bn3v = (const float*)d_in[15];
  const float* W4  = (const float*)d_in[16];
  const float* bn4g = (const float*)d_in[17], *bn4b = (const float*)d_in[18];
  const float* bn4m = (const float*)d_in[19], *bn4v = (const float*)d_in[20];
  const float* W5  = (const float*)d_in[21];
  const float* bn5g = (const float*)d_in[22], *bn5b = (const float*)d_in[23];
  const float* bn5m = (const float*)d_in[24], *bn5v = (const float*)d_in[25];
  const float* Ws1 = (const float*)d_in[26];
  const float* bs1g = (const float*)d_in[27], *bs1b = (const float*)d_in[28];
  const float* bs1m = (const float*)d_in[29], *bs1v = (const float*)d_in[30];
  const float* Ws2 = (const float*)d_in[31];
  const float* bs2g = (const float*)d_in[32], *bs2b = (const float*)d_in[33];
  const float* bs2m = (const float*)d_in[34], *bs2v = (const float*)d_in[35];
  const float* Ws3 = (const float*)d_in[36];
  float* out = (float*)d_out;

  // ---- workspace carve-up ----
  char* ws = (char*)d_ws;
  size_t off = 0;
  auto carve = [&](size_t bytes) {
    void* p = ws + off;
    off = (off + bytes + 255) & ~(size_t)255;
    return p;
  };
  const size_t M = (size_t)BATCH * NPTS;                   // 16384 rows
  int*      idx   = (int*)      carve(M * KNN * sizeof(int));
  float*    cat   = (float*)    carve(M * 512 * sizeof(float));    // fp32 for knn
  _Float16* cath  = (_Float16*) carve(M * 512 * sizeof(_Float16)); // f16 for GEMM
  _Float16* h1h   = (_Float16*) carve(M * 512 * sizeof(_Float16));
  _Float16* h2h   = (_Float16*) carve(M * 256 * sizeof(_Float16));
  float*    gmax  = (float*)    carve(BATCH * 1024 * sizeof(float));
  _Float16* gmaxh = (_Float16*) carve(BATCH * 1024 * sizeof(_Float16));
  _Float16* W1p  = (_Float16*)carve(packed_halfs(14,   64)  * 2);
  _Float16* W2p  = (_Float16*)carve(packed_halfs(128,  64)  * 2);
  _Float16* W3p  = (_Float16*)carve(packed_halfs(128,  128) * 2);
  _Float16* W4p  = (_Float16*)carve(packed_halfs(256,  256) * 2);
  _Float16* W5p  = (_Float16*)carve(packed_halfs(512,  1024)* 2);
  _Float16* Ws1p = (_Float16*)carve(packed_halfs(1536, 512) * 2);
  _Float16* Ws2p = (_Float16*)carve(packed_halfs(512,  256) * 2);
  _Float16* Ws3p = (_Float16*)carve(packed_halfs(256,  13)  * 2);
  (void)ws_size; (void)in_sizes; (void)n_in; (void)out_size;

  auto pack = [&](const float* W, _Float16* Wp, int K, int N) {
    int KT = (K + 31) / 32, NT = (N + 15) / 16;
    int threads = NT * KT * 32;
    pack_w_kernel<<<(threads + 255) / 256, 256, 0, stream>>>(W, Wp, K, N, KT, NT);
  };
  pack(W1,  W1p,  14,   64);
  pack(W2,  W2p,  128,  64);
  pack(W3,  W3p,  128,  128);
  pack(W4,  W4p,  256,  256);
  pack(W5,  W5p,  512,  1024);
  pack(Ws1, Ws1p, 1536, 512);
  pack(Ws2, Ws2p, 512,  256);
  pack(Ws3, Ws3p, 256,  13);

  const int nPts = BATCH * NPTS;
  const dim3 ecGrid(nPts / 4), ecBlk(128);

  // EdgeConv 1: x(7) -> cat[:,0:64]
  knn_kernel<7><<<(nPts + 255) / 256, 256, 0, stream>>>(x, 7, idx);
  edgeconv_kernel<7, 64><<<ecGrid, ecBlk, 0, stream>>>(
      x, 7, idx, W1p, bn1g, bn1b, bn1m, bn1v, cat, cath, 512, 0);
  // EdgeConv 2: x1(64) -> cat[:,64:128]
  knn_kernel<64><<<(nPts + 255) / 256, 256, 0, stream>>>(cat, 512, idx);
  edgeconv_kernel<64, 64><<<ecGrid, ecBlk, 0, stream>>>(
      cat, 512, idx, W2p, bn2g, bn2b, bn2m, bn2v, cat, cath, 512, 64);
  // EdgeConv 3: x2(64) -> cat[:,128:256]
  knn_kernel<64><<<(nPts + 255) / 256, 256, 0, stream>>>(cat + 64, 512, idx);
  edgeconv_kernel<64, 128><<<ecGrid, ecBlk, 0, stream>>>(
      cat + 64, 512, idx, W3p, bn3g, bn3b, bn3m, bn3v, cat, cath, 512, 128);
  // EdgeConv 4: x3(128) -> cat[:,256:512]
  knn_kernel<128><<<(nPts + 255) / 256, 256, 0, stream>>>(cat + 128, 512, idx);
  edgeconv_kernel<128, 256><<<ecGrid, ecBlk, 0, stream>>>(
      cat + 128, 512, idx, W4p, bn4g, bn4b, bn4m, bn4v, cat, cath, 512, 256);

  const int Mtiles = (int)(M / 16);   // 1024

  // g = lrelu(bn5(cat@W5)); gmax = max over N (fused, no 67MB intermediate)
  fill_kernel<<<(BATCH * 1024 + 255) / 256, 256, 0, stream>>>(gmax, -3.4e38f,
                                                              BATCH * 1024);
  gemm_kernel<true, true, true, false>
      <<<dim3(Mtiles, 64 / 4), 128, 16 * 512 * 2, stream>>>(
      cath, 512, nullptr, 0, /*splitK=*/512, W5p, 512, 64, 1024,
      bn5g, bn5b, bn5m, bn5v, gmax, 1024);
  f32_to_f16_kernel<<<(BATCH * 1024 + 255) / 256, 256, 0, stream>>>(
      gmax, gmaxh, BATCH * 1024);

  // h1 = lrelu(bns1([cat | gmax] @ Ws1)), K=1536 virtual concat
  gemm_kernel<true, true, false, true>
      <<<dim3(Mtiles, 32 / 4), 128, 16 * 1536 * 2, stream>>>(
      cath, 512, gmaxh, 1024, /*splitK=*/512, Ws1p, 1536, 32, 512,
      bs1g, bs1b, bs1m, bs1v, h1h, 512);

  // h2 = lrelu(bns2(h1 @ Ws2))
  gemm_kernel<true, true, false, true>
      <<<dim3(Mtiles, 16 / 4), 128, 16 * 512 * 2, stream>>>(
      h1h, 512, nullptr, 0, /*splitK=*/512, Ws2p, 512, 16, 256,
      bs2g, bs2b, bs2m, bs2v, h2h, 256);

  // out = h2 @ Ws3  (N=13, padded N-tile with guarded stores)
  gemm_kernel<false, false, false, false>
      <<<dim3(Mtiles, 1), 128, 16 * 256 * 2, stream>>>(
      h2h, 256, nullptr, 0, /*splitK=*/256, Ws3p, 256, 1, 13,
      nullptr, nullptr, nullptr, nullptr, out, 13);
}
